// SurrogateNet2_7808250544423
// MI455X (gfx1250) — compile-verified, matrix operand order checked by reading, not run on hardware
//
#include <hip/hip_runtime.h>
#include <hip/hip_bf16.h>
#include <math.h>

// ---------------------------------------------------------------------------
// SurrogateNet: warp-STN + conv chain, fused for MI455X (gfx1250, wave32).
// All significant convs (loc conv1: 1.4 GFLOP, n1c2: 17 GFLOP, n1t1: 4.3
// GFLOP) run on v_wmma_f32_16x16x32_f16 as implicit GEMMs fed by
// ds_load_b128; the rest is fused VALU through LDS.
// ---------------------------------------------------------------------------

#define BATCH   256
#define SEQL    140
#define CHN     93
#define NTOT    (SEQL * CHN)          // 13020
#define POFF    (BATCH * NTOT)        // param offset in d_out
#define SOFF    (POFF + BATCH)        // scale offset in d_out

typedef __attribute__((ext_vector_type(16))) _Float16 v16h;
typedef __attribute__((ext_vector_type(8)))  float    v8f;
typedef __attribute__((ext_vector_type(4)))  unsigned int u32x4;

union V16U { v16h h; u32x4 u[2]; };

// ---------------------------------------------------------------------------
// Kernel A: localization net. One workgroup (256 thr = 8 waves) per batch.
// conv1 (93->32, K7) via WMMA: M=pos (9 tiles), N=16x2 passes,
// K = 7 taps x 96 (93 zero-padded) = 672 = 21 k-steps of 32.
// ReLU+maxpool fused in the WMMA epilogue. conv2 + dots stay VALU (tiny).
// Writes param/scale to d_out tail.
// ---------------------------------------------------------------------------
__global__ __launch_bounds__(256)
void loc_kernel(const float* __restrict__ x,
                const float* __restrict__ w1, const float* __restrict__ b1,
                const float* __restrict__ w2, const float* __restrict__ b2,
                const float* __restrict__ shw, const float* __restrict__ shb,
                const float* __restrict__ scw, const float* __restrict__ scb,
                float* __restrict__ out)
{
    __shared__ __align__(16) _Float16 xf[150 * 96];   // 28800 B, zero-padded
    __shared__ __align__(16) _Float16 Bl[16 * 672];   // 21504 B, per-pass B
    __shared__ __align__(16) float pool1[32 * 67];    //  8576 B
    __shared__ __align__(16) float red[512];          //  2048 B

    const int tid  = threadIdx.x;
    const int lane = tid & 31;
    const int wave = tid >> 5;          // 0..7
    const int bi   = blockIdx.x;
    const float* xb = x + bi * NTOT;

    // stage x as f16 [row][96]; rows >=140 and cols >=93 are zero (halo/K pad)
    for (int i = tid; i < 150 * 96; i += 256) {
        int l = i / 96, c = i % 96;
        float v = (l < SEQL && c < CHN) ? xb[l * CHN + c] : 0.f;
        xf[i] = (_Float16)v;
    }

    const int Ncol  = lane & 15;
    const int mrow  = lane & 15;
    const int mbase = (lane < 16) ? 0 : 8;
    const int hoff  = (lane < 16) ? 0 : 16;          // byte half-offset

    for (int g = 0; g < 2; ++g) {                    // two 16-channel passes
        __syncthreads();                             // Bl safe to overwrite
        // B image: Bl[o][tap*96 + c] = w1[(o+16g)][c][tap], zero-padded c
        for (int i = tid; i < 16 * 672; i += 256) {
            int o = i / 672, r = i % 672;
            int tap = r / 96, c = r % 96;
            float v = (c < CHN) ? w1[(o + 16 * g) * 651 + c * 7 + tap] : 0.f;
            Bl[i] = (_Float16)v;
        }
        __syncthreads();
        const float b1o = b1[Ncol + 16 * g];

        for (int t = wave; t < 9; t += 8) {          // wave-uniform M tiles
            int pbase = 16 * t;
            v8f acc = (v8f)0.f;
            #pragma unroll
            for (int s = 0; s < 21; ++s) {
                int tap = s / 3, sub = s % 3;
                V16U A, Bv;
                const char* ab = (const char*)xf
                    + ((pbase + mrow + tap) * 96 + sub * 32) * 2 + hoff;
                A.u[0] = *(const u32x4*)(ab);
                A.u[1] = *(const u32x4*)(ab + 32);
                const char* bb = (const char*)Bl
                    + (Ncol * 672 + tap * 96 + sub * 32) * 2 + hoff;
                Bv.u[0] = *(const u32x4*)(bb);
                Bv.u[1] = *(const u32x4*)(bb + 32);
                acc = __builtin_amdgcn_wmma_f32_16x16x32_f16(
                          false, A.h, false, Bv.h, (short)0, acc, false, false);
            }
            // fused bias + relu + maxpool2: pairs (r, r+1) live in-lane
            #pragma unroll
            for (int r = 0; r < 8; r += 2) {
                int pos0 = pbase + mbase + r;        // even
                if (pos0 < 134) {
                    float v = fmaxf(fmaxf(acc[r], acc[r + 1]) + b1o, 0.f);
                    pool1[(Ncol + 16 * g) * 67 + (pos0 >> 1)] = v;
                }
            }
        }
    }
    __syncthreads();

    // conv2 (32->64, K5, valid) + maxpool2 + relu, fused with dot products
    float acc_sh = 0.f, acc_sc = 0.f;
    for (int f = tid; f < 64 * 31; f += 256) {
        int o = f / 31, l2 = f % 31;
        int l0 = 2 * l2, l1 = l0 + 1;
        float a0 = b2[o], a1 = b2[o];
        for (int ci = 0; ci < 32; ++ci) {
            const float* pr = &pool1[ci * 67];
            const float* wr = &w2[o * 160 + ci * 5];
            #pragma unroll
            for (int k = 0; k < 5; ++k) {
                float wv = wr[k];
                a0 += wv * pr[l0 + k];
                a1 += wv * pr[l1 + k];
            }
        }
        float v = fmaxf(fmaxf(a0, a1), 0.f);
        acc_sh += v * shw[f];
        acc_sc += v * scw[f];
    }
    red[tid]       = acc_sh;
    red[256 + tid] = acc_sc;
    __syncthreads();
    for (int off = 128; off > 0; off >>= 1) {
        if (tid < off) {
            red[tid]       += red[tid + off];
            red[256 + tid] += red[256 + tid + off];
        }
        __syncthreads();
    }
    if (tid == 0) {
        float pv = tanhf(red[0] + shb[0]);
        pv = fminf(fmaxf(pv, -1.f + 1e-8f), 1.f - 1e-8f) * 15.f;
        float sv = 1.f + tanhf(red[256] + scb[0]) * 0.25f;
        out[POFF + bi] = pv;
        out[SOFF + bi] = sv;
    }
}

// ---------------------------------------------------------------------------
// Kernel C: warp + n1c1 + n1c2(WMMA) + n1t1(WMMA) + n1t2 + n1c3 + sigmoid.
// Grid: (51 tiles of 256 positions, 256 batch). 128 threads = 4 waves.
// Local window: 288 positions, origin LO = n0 - 16.
//   x2s valid global range [0,13019]; y1 [0,13015]; y2 [0,13011]; t1 [0,13015]
// ---------------------------------------------------------------------------
__global__ __launch_bounds__(128)
void chain_kernel(const float* __restrict__ x,
                  const float* __restrict__ c1w, const float* __restrict__ c1b,
                  const float* __restrict__ c2w, const float* __restrict__ c2b,
                  const float* __restrict__ t1w, const float* __restrict__ t1b,
                  const float* __restrict__ t2w, const float* __restrict__ t2b,
                  const float* __restrict__ c3w, const float* __restrict__ c3b,
                  float* __restrict__ out)
{
    __shared__ __align__(16) float    x2s[288];          //  1152 B
    __shared__ __align__(16) _Float16 y1s[288 * 32];     // 18432 B
    __shared__ __align__(16) _Float16 y2s[288 * 16];     //  9216 B
    __shared__ __align__(16) _Float16 t1s[288 * 8];      //  4608 B

    const int tid  = threadIdx.x;
    const int lane = tid & 31;
    const int wave = tid >> 5;
    const int bi   = blockIdx.y;
    const int n0   = blockIdx.x * 256;
    const int LO   = n0 - 16;
    const float* xg = x + bi * NTOT;

    const float shiftv = out[POFF + bi];  // written by loc_kernel (stream order)
    const float scalev = out[SOFF + bi];

    // ---- Phase 0: warp (gamma lerp) into x2s; zero-init y2s/t1s ----
    for (int p = tid; p < 288; p += 128) {
        int g = LO + p;
        float v = 0.f;
        if (g >= 0 && g < NTOT) {
            int s = g / CHN;
            int c = g - s * CHN;
            float gm;
            if (s == 0)            gm = 0.f;
            else if (s == SEQL-1)  gm = (float)SEQL;
            else {
                gm = scalev * (float)s + shiftv;
                gm = fminf(fmaxf(gm, 0.f), (float)SEQL);
            }
            float g0f = fminf(fmaxf(floorf(gm), 0.f), (float)(SEQL - 1));
            int g0 = (int)g0f;
            int g1 = g0 + 1; if (g1 > SEQL - 1) g1 = SEQL - 1;
            float wa = 1.f - (gm - g0f);
            v = wa * xg[g0 * CHN + c] + (1.f - wa) * xg[g1 * CHN + c];
        }
        x2s[p] = v;
    }
    for (int i = tid; i < 288 * 16; i += 128) y2s[i] = (_Float16)0.f;
    for (int i = tid; i < 288 * 8;  i += 128) t1s[i] = (_Float16)0.f;

    // ---- Per-lane B operands (kept in VGPRs for the whole block) ----
    const int Ncol  = lane & 15;                 // output channel / column
    const int cbase = (lane < 16) ? 0 : 8;       // 16-bit A/B lane K-base
    v16h Bw2[5];
    #pragma unroll
    for (int s = 0; s < 5; ++s) {                // n1c2: K = tap*32 + ch
        v16h bv;
        #pragma unroll
        for (int e = 0; e < 16; ++e) {
            int c = cbase + (e & 7) + ((e >> 3) << 4);
            bv[e] = (_Float16)c2w[Ncol * 160 + c * 5 + s];
        }
        Bw2[s] = bv;
    }
    v16h Bt1[3];
    #pragma unroll
    for (int s = 0; s < 3; ++s) {                // n1t1: K = tap*16 + ch (pad 96)
        v16h bv;
        #pragma unroll
        for (int e = 0; e < 16; ++e) {
            int Kl = cbase + (e & 7) + ((e >> 3) << 4);
            int K  = s * 32 + Kl;
            int kp = K >> 4, c = K & 15;
            float vv = 0.f;
            if (Ncol < 8 && kp < 5)
                vv = t1w[c * 40 + Ncol * 5 + (4 - kp)];  // (in=16,out=8,K) flipped
            bv[e] = (_Float16)vv;
        }
        Bt1[s] = bv;
    }
    const float bias2 = c2b[Ncol];
    const float biasT = (Ncol < 8) ? t1b[Ncol] : 0.f;
    __syncthreads();

    // ---- Phase 1: n1c1 (1->32, K5) + relu -> y1s[pos][32] f16 ----
    for (int idx = tid; idx < 288 * 32; idx += 128) {
        int p = idx >> 5, c = idx & 31;
        int g = LO + p;
        float v = 0.f;
        if (p <= 283 && g >= 0 && g <= NTOT - 5) {
            v = c1b[c];
            #pragma unroll
            for (int k = 0; k < 5; ++k) v += c1w[c * 5 + k] * x2s[p + k];
            v = fmaxf(v, 0.f);
        }
        y1s[p * 32 + c] = (_Float16)v;
    }
    __syncthreads();

    // ---- Phase 2: n1c2 via WMMA (M=pos, N=16, K=160 in 5 steps of 32) ----
    const int mrow  = lane & 15;
    const int mbase = (lane < 16) ? 0 : 8;
    const int hoff  = (lane < 16) ? 0 : 16;      // byte offset into 64B y1 row
    for (int t = wave; t < 17; t += 4) {         // wave-uniform: EXEC stays full
        int pbase = 8 + 16 * t;
        int arow  = pbase + mrow;
        v8f acc = (v8f)0.f;
        #pragma unroll
        for (int s = 0; s < 5; ++s) {
            V16U A;
            const char* base = (const char*)y1s + (arow + s) * 64 + hoff;
            A.u[0] = *(const u32x4*)(base);
            A.u[1] = *(const u32x4*)(base + 32);
            acc = __builtin_amdgcn_wmma_f32_16x16x32_f16(
                      false, A.h, false, Bw2[s], (short)0, acc, false, false);
        }
        #pragma unroll
        for (int r = 0; r < 8; ++r) {
            int row = pbase + mbase + r;
            int j   = LO + row;
            float v = fmaxf(acc[r] + bias2, 0.f);
            if (j < 0 || j > NTOT - 9) v = 0.f;          // y2 valid [0,13011]
            y2s[row * 16 + Ncol] = (_Float16)v;
        }
    }
    __syncthreads();

    // ---- Phase 3: n1t1 (convT 16->8) via WMMA (K=80 padded to 96) + tanh ----
    for (int t = wave; t < 17; t += 4) {
        int pbase = 12 + 16 * t;
        int arow  = pbase + mrow;
        v8f acc = (v8f)0.f;
        #pragma unroll
        for (int s = 0; s < 3; ++s) {
            V16U A;
            int r0 = arow - 4 + 2 * s;
            const char* base = (const char*)y2s + r0 * 32 + hoff;
            A.u[0] = *(const u32x4*)(base);
            if (s < 2) A.u[1] = *(const u32x4*)(base + 32);
            else       A.u[1] = (u32x4)0u;               // K = 80..95 zero pad
            acc = __builtin_amdgcn_wmma_f32_16x16x32_f16(
                      false, A.h, false, Bt1[s], (short)0, acc, false, false);
        }
        if (Ncol < 8) {
            #pragma unroll
            for (int r = 0; r < 8; ++r) {
                int row = pbase + mbase + r;
                int m   = LO + row;
                float v = tanhf(acc[r] + biasT);
                if (m < 0 || m > NTOT - 5) v = 0.f;      // t1 valid [0,13015]
                t1s[row * 8 + Ncol] = (_Float16)v;
            }
        }
    }
    __syncthreads();

    // ---- Phase 4: n1t2 (convT 8->1) + tanh + n1c3 + sigmoid + residual ----
    float wt2r[5][8];
    #pragma unroll
    for (int kp = 0; kp < 5; ++kp)
        #pragma unroll
        for (int c = 0; c < 8; ++c)
            wt2r[kp][c] = t2w[c * 5 + (4 - kp)];         // (8,1,5) flipped
    const float tb2 = t2b[0];
    const float w3v = c3w[0], b3v = c3b[0];

    #pragma unroll
    for (int rep = 0; rep < 2; ++rep) {
        int idx = rep * 128 + tid;
        int n   = n0 + idx;
        if (n < NTOT) {
            int p = 16 + idx;
            float a = tb2;
            #pragma unroll
            for (int kp = 0; kp < 5; ++kp) {
                const _Float16* tr = &t1s[(p - 4 + kp) * 8];
                #pragma unroll
                for (int c = 0; c < 8; ++c) a += (float)tr[c] * wt2r[kp][c];
            }
            float t2v = tanhf(a);
            float z   = w3v * t2v + b3v + x2s[p];
            float sg  = 1.f / (1.f + expf(-z));
            sg = fminf(fmaxf(sg, 1e-8f), 1.f - 1e-8f);
            out[bi * NTOT + n] = sg;
        }
    }
}

// ---------------------------------------------------------------------------
extern "C" void kernel_launch(void* const* d_in, const int* in_sizes, int n_in,
                              void* d_out, int out_size, void* d_ws, size_t ws_size,
                              hipStream_t stream)
{
    const float* x    = (const float*)d_in[0];
    const float* l1w  = (const float*)d_in[1];
    const float* l1b  = (const float*)d_in[2];
    const float* l2w  = (const float*)d_in[3];
    const float* l2b  = (const float*)d_in[4];
    const float* shw  = (const float*)d_in[5];
    const float* shb  = (const float*)d_in[6];
    const float* scw  = (const float*)d_in[7];
    const float* scb  = (const float*)d_in[8];
    const float* c1w  = (const float*)d_in[9];
    const float* c1b  = (const float*)d_in[10];
    const float* c2w  = (const float*)d_in[11];
    const float* c2b  = (const float*)d_in[12];
    const float* t1w  = (const float*)d_in[13];
    const float* t1b  = (const float*)d_in[14];
    const float* t2w  = (const float*)d_in[15];
    const float* t2b  = (const float*)d_in[16];
    const float* c3w  = (const float*)d_in[17];
    const float* c3b  = (const float*)d_in[18];
    float* out = (float*)d_out;

    loc_kernel<<<dim3(BATCH), dim3(256), 0, stream>>>(
        x, l1w, l1b, l2w, l2b, shw, shb, scw, scb, out);

    chain_kernel<<<dim3(51, BATCH), dim3(128), 0, stream>>>(
        x, c1w, c1b, c2w, c2b, t1w, t1b, t2w, t2b, c3w, c3b, out);
}
